// CNNAttention_77979426226593
// MI455X (gfx1250) — matrix operand order repaired and checked by reading, not compiled, not run on hardware
//
#include <hip/hip_runtime.h>
#include <hip/hip_bf16.h>

// ---------------------------------------------------------------------------
// CNN self-attention, MI455X (gfx1250), wave32 + WMMA bf16 (f32 accumulate).
// B=16, C=256, H=W=64 -> N=4096, CQK=32.
//
// Pipeline:
//   1) pack_kernel : wq/wk/wv -> fused bf16 weight [320][256] + bias[320]
//   2) proj_kernel : Y[b][320][4096] = W @ x[b] + bias   (WMMA bf16 GEMM)
//   3) attn_kernel : flash-attention over Y; staging loads for chunk i+1
//                    overlap compute of chunk i. Barriers wait ONLY on DScnt
//                    (split CDNA5 counters) so in-flight global prefetches
//                    are not drained at the barrier.
// ---------------------------------------------------------------------------

#define BN    16
#define CCH   256
#define NTOK  4096
#define OALL  320   // 32 q + 32 k + 256 v

typedef __attribute__((ext_vector_type(16))) __bf16 v16bf;
typedef __attribute__((ext_vector_type(8)))  float  v8f;

union FragBF { v16bf v; uint4 u[2]; };

// Workgroup barrier that drains only the LDS counter: outstanding global
// loads (private VGPR destinations) are deliberately left in flight.
__device__ __forceinline__ void wg_barrier_ds() {
  asm volatile("s_wait_dscnt 0\n\t"
               "s_barrier_signal -1\n\t"
               "s_barrier_wait -1" ::: "memory");
}

__device__ __forceinline__ unsigned short f2bf(float f) {
  unsigned int u = __float_as_uint(f);
  u += 0x7FFFu + ((u >> 16) & 1u);          // round-to-nearest-even
  return (unsigned short)(u >> 16);
}

__device__ __forceinline__ v8f v8f_zero() {
  v8f z;
#pragma unroll
  for (int r = 0; r < 8; ++r) z[r] = 0.0f;
  return z;
}

// A-matrix 16x32 bf16 fragment (ISA 7.12.2):
// lane<16: M=lane,    K = {0..7} in e0..7,  {16..23} in e8..15
// lane>=16:M=lane-16, K = {8..15} in e0..7, {24..31} in e8..15
__device__ __forceinline__ v16bf load_a_frag(const unsigned short* rowBase, int half) {
  FragBF f;
  f.u[0] = *reinterpret_cast<const uint4*>(rowBase + half * 8);
  f.u[1] = *reinterpret_cast<const uint4*>(rowBase + 16 + half * 8);
  return f.v;
}

// B-matrix 32x16 bf16 fragment: lane<16 column N=lane holds K={0..7,16..23},
// lane>=16 column N=lane-16 holds K={8..15,24..31} (same order as A).
__device__ __forceinline__ v16bf load_b_frag(const unsigned short* rowBase, int half) {
  FragBF f;
  f.u[0] = *reinterpret_cast<const uint4*>(rowBase + half * 8);
  f.u[1] = *reinterpret_cast<const uint4*>(rowBase + 16 + half * 8);
  return f.v;
}

__device__ __forceinline__ v8f wmma_bf16(v16bf a, v16bf b, v8f c) {
  return __builtin_amdgcn_wmma_f32_16x16x32_bf16(false, a, false, b, (short)0, c,
                                                 false, false);
}

__device__ __forceinline__ unsigned int ld_u32(const char* base, unsigned int offB) {
  return *reinterpret_cast<const unsigned int*>(base + offB);
}
__device__ __forceinline__ float ld_f32(const char* base, unsigned int offB) {
  return *reinterpret_cast<const float*>(base + offB);
}
__device__ __forceinline__ unsigned int ld_u16(const char* base, unsigned int offB) {
  return *reinterpret_cast<const unsigned short*>(base + offB);
}

// ---------------------------------------------------------------------------
// 1) pack weights: Wall[o][c] bf16, biasAll[o]
// ---------------------------------------------------------------------------
__global__ void pack_kernel(const float* __restrict__ wq, const float* __restrict__ bq,
                            const float* __restrict__ wk, const float* __restrict__ bk,
                            const float* __restrict__ wv, const float* __restrict__ bv,
                            unsigned short* __restrict__ Wall, float* __restrict__ biasAll) {
  int o = blockIdx.x, c = threadIdx.x;
  float w;
  if (o < 32)       w = wq[o * CCH + c];
  else if (o < 64)  w = wk[(o - 32) * CCH + c];
  else              w = wv[(o - 64) * CCH + c];
  Wall[o * CCH + c] = f2bf(w);
  if (c == 0)
    biasAll[o] = (o < 32) ? bq[o] : (o < 64) ? bk[o - 32] : bv[o - 64];
}

// ---------------------------------------------------------------------------
// 2) projection GEMM: Y[b][o][n] = sum_c Wall[o][c]*x[b][c][n] + bias[o]
//    WG = 256 threads (8 waves), tile 64(o) x 64(n), K-loop 8 x 32,
//    global loads for step s+1 issued before the WMMAs of step s.
// ---------------------------------------------------------------------------
__global__ __launch_bounds__(256) void proj_kernel(const float* __restrict__ x,
                                                   const unsigned short* __restrict__ Wall,
                                                   const float* __restrict__ biasAll,
                                                   unsigned short* __restrict__ Y) {
  __shared__ __align__(16) unsigned short ldsW[64][40];  // [o][k]  (pad 40)
  __shared__ __align__(16) unsigned short ldsX[64][40];  // [n][k]

  const int tid  = threadIdx.x;
  const int wid  = tid >> 5, lane = tid & 31;
  const int half = lane >> 4, l16 = lane & 15;
  const int b  = blockIdx.z;
  const int o0 = blockIdx.y * 64;
  const int n0 = blockIdx.x * 64;
  const int ot = (wid & 3) * 16;     // wave's o sub-tile
  const int nt = (wid >> 2) * 32;    // wave's n sub-tile (two 16-wide halves)

  // loop-invariant per-lane byte offsets; uniform base advances each K step
  unsigned int woffB[8], xoffB[8];
#pragma unroll
  for (int j = 0; j < 8; ++j) {
    int idx = tid + 256 * j;
    int o = idx >> 5, k = idx & 31;
    woffB[j] = (unsigned int)(((o0 + o) * CCH + k) * 2);
  }
#pragma unroll
  for (int j = 0; j < 8; ++j) {
    int idx = tid + 256 * j;
    int k = idx >> 6, nl = idx & 63;
    xoffB[j] = (unsigned int)((k * NTOK + nl) * 4);
  }
  const char* baseX0 = (const char*)(x + (size_t)b * CCH * NTOK + n0);

  unsigned int wreg[8];
  float        xreg[8];

  auto issue_loads = [&](int k0) {
    const char* bw = (const char*)Wall + (size_t)k0 * 2;
    const char* bx = baseX0 + (size_t)k0 * NTOK * 4;
#pragma unroll
    for (int j = 0; j < 8; ++j) wreg[j] = ld_u16(bw, woffB[j]);
#pragma unroll
    for (int j = 0; j < 8; ++j) xreg[j] = ld_f32(bx, xoffB[j]);
  };
  auto drain_to_lds = [&]() {
#pragma unroll
    for (int j = 0; j < 8; ++j) {
      int idx = tid + 256 * j;
      int o = idx >> 5, k = idx & 31;
      ldsW[o][k] = (unsigned short)wreg[j];
    }
#pragma unroll
    for (int j = 0; j < 8; ++j) {
      int idx = tid + 256 * j;
      int k = idx >> 6, nl = idx & 63;
      ldsX[nl][k] = f2bf(xreg[j]);
    }
  };

  v8f acc0 = v8f_zero(), acc1 = v8f_zero();

  issue_loads(0);
  for (int s = 0; s < 8; ++s) {
    drain_to_lds();
    wg_barrier_ds();
    if (s < 7) issue_loads((s + 1) * 32);    // in flight across the WMMAs
                                             // and the trailing barrier
    v16bf aw = load_a_frag(ldsW[ot + l16], half);
    v16bf b0 = load_b_frag(ldsX[nt + l16], half);
    v16bf b1 = load_b_frag(ldsX[nt + 16 + l16], half);
    acc0 = wmma_bf16(aw, b0, acc0);
    acc1 = wmma_bf16(aw, b1, acc1);
    wg_barrier_ds();
  }

#pragma unroll
  for (int r = 0; r < 8; ++r) {
    int   o  = o0 + ot + r + half * 8;
    float bv = biasAll[o];
    int   na = n0 + nt + l16;
    size_t base = ((size_t)b * OALL + o) * NTOK;
    Y[base + na]      = f2bf(acc0[r] + bv);
    Y[base + na + 16] = f2bf(acc1[r] + bv);
  }
}

// ---------------------------------------------------------------------------
// 3) fused flash attention.
//    WG = 256 threads (8 waves) per 128 query tokens; wave owns 16 rows.
//    KV chunk = 32; per chunk: 2 score WMMAs + 16 PV WMMAs per wave.
// ---------------------------------------------------------------------------
__global__ __launch_bounds__(256) void attn_kernel(const unsigned short* __restrict__ Y,
                                                   const float* __restrict__ x,
                                                   const float* __restrict__ gamma,
                                                   float* __restrict__ out) {
  // phase-1 layout (bytes): Q 10240 | K 2560 | Vt 20480 | P 10240  = 43520
  // phase-2 overlays ldsO (32x132 f32 = 16896) at offset 0.
  __shared__ __align__(16) unsigned char smem[43520];
  unsigned short (*ldsQ)[40] = (unsigned short (*)[40])(smem);           // [128 tok][32 d]
  unsigned short (*ldsK)[40] = (unsigned short (*)[40])(smem + 10240);   // [32 kv][32 d]
  unsigned short (*ldsV)[40] = (unsigned short (*)[40])(smem + 12800);   // [256 c][32 kv]
  unsigned short (*ldsP)[40] = (unsigned short (*)[40])(smem + 33280);   // [8*16][32 kv]
  float (*ldsO)[132]         = (float (*)[132])(smem);                   // [32 c][128 n]

  const int tid  = threadIdx.x;
  const int wid  = tid >> 5, lane = tid & 31;
  const int half = lane >> 4, l16 = lane & 15;
  const int b  = blockIdx.y;
  const int n0 = blockIdx.x * 128;
  const unsigned short* Yb = Y + (size_t)b * OALL * NTOK;

  // stage this block's q rows: ldsQ[nl][d] = Y[b][d][n0+nl]
#pragma unroll
  for (int j = 0; j < 16; ++j) {
    int idx = tid + 256 * j;               // 128*32
    int d = idx >> 7, nl = idx & 127;
    ldsQ[nl][d] = Yb[(size_t)d * NTOK + n0 + nl];
  }
  wg_barrier_ds();

  const v16bf qa = load_a_frag(ldsQ[wid * 16 + l16], half);   // invariant A frag

  // loop-invariant 32-bit byte offsets for staging; base = Yb + kv0 advances
  unsigned int koffB[2], voffB[16];
#pragma unroll
  for (int j = 0; j < 2; ++j) {            // 32x32 K chunk as dwords
    int idx = tid + 256 * j;               // 512 dwords
    int d = idx >> 4, kp = idx & 15;
    koffB[j] = (unsigned int)(((32 + d) * NTOK + 2 * kp) * 2);
  }
#pragma unroll
  for (int j = 0; j < 16; ++j) {           // 256x32 V^T chunk as dwords
    int idx = tid + 256 * j;
    int c = idx >> 4, kp = idx & 15;
    voffB[j] = (unsigned int)(((64 + c) * NTOK + 2 * kp) * 2);
  }

  unsigned int kreg[2];
  unsigned int vreg[16];
  auto issue_loads = [&](int kv0) {
    const char* base = (const char*)(Yb + kv0);
#pragma unroll
    for (int j = 0; j < 2; ++j)  kreg[j] = ld_u32(base, koffB[j]);
#pragma unroll
    for (int j = 0; j < 16; ++j) vreg[j] = ld_u32(base, voffB[j]);
  };
  auto drain_to_lds = [&]() {
#pragma unroll
    for (int j = 0; j < 2; ++j) {
      int idx = tid + 256 * j;
      int d = idx >> 4, kp = idx & 15;
      ldsK[2 * kp][d]     = (unsigned short)kreg[j];
      ldsK[2 * kp + 1][d] = (unsigned short)(kreg[j] >> 16);
    }
#pragma unroll
    for (int j = 0; j < 16; ++j) {
      int idx = tid + 256 * j;
      int c = idx >> 4, kp = idx & 15;
      *reinterpret_cast<unsigned int*>(&ldsV[c][2 * kp]) = vreg[j];
    }
  };

  float m[8], ssum[8];
  v8f   acc[16];
#pragma unroll
  for (int r = 0; r < 8; ++r) { m[r] = -3.0e38f; ssum[r] = 0.0f; }
#pragma unroll
  for (int t = 0; t < 16; ++t) acc[t] = v8f_zero();

  issue_loads(0);
  for (int kv0 = 0; kv0 < NTOK; kv0 += 32) {
    drain_to_lds();                        // s_wait_loadcnt lands here, one
    wg_barrier_ds();                       // full chunk after issue
    if (kv0 + 32 < NTOK) issue_loads(kv0 + 32);

    // scores: two 16x16 tiles over this 32-key chunk
    v16bf kb0 = load_b_frag(ldsK[l16], half);
    v16bf kb1 = load_b_frag(ldsK[16 + l16], half);
    v8f s0 = wmma_bf16(qa, kb0, v8f_zero());
    v8f s1 = wmma_bf16(qa, kb1, v8f_zero());

    // online softmax (row reductions across the 16-lane half-groups)
    float al[8];
    bool  grew = false;
#pragma unroll
    for (int r = 0; r < 8; ++r) {
      float v = fmaxf(s0[r], s1[r]);
      v = fmaxf(v, __shfl_xor(v, 1, 32));
      v = fmaxf(v, __shfl_xor(v, 2, 32));
      v = fmaxf(v, __shfl_xor(v, 4, 32));
      v = fmaxf(v, __shfl_xor(v, 8, 32));
      float mo = m[r];
      float mn = fmaxf(mo, v);
      al[r] = __expf(mo - mn);
      grew  = grew || (mn > mo);
      m[r]  = mn;
      float p0 = __expf(s0[r] - mn);
      float p1 = __expf(s1[r] - mn);
      s0[r] = p0; s1[r] = p1;
      float ps = p0 + p1;
      ps += __shfl_xor(ps, 1, 32);
      ps += __shfl_xor(ps, 2, 32);
      ps += __shfl_xor(ps, 4, 32);
      ps += __shfl_xor(ps, 8, 32);
      ssum[r] = ssum[r] * al[r] + ps;
    }
    // O-rescale only when some row's max actually increased (alpha != 1)
    if (__any((int)grew)) {
#pragma unroll
      for (int t = 0; t < 16; ++t) {
#pragma unroll
        for (int r = 0; r < 8; ++r) acc[t][r] *= al[r];
      }
    }

    // D-layout P -> A-layout via per-wave LDS (same-wave DS ops are in order)
    unsigned short (*pw)[40] = &ldsP[wid * 16];
#pragma unroll
    for (int r = 0; r < 8; ++r) {
      pw[r + half * 8][l16]      = f2bf(s0[r]);
      pw[r + half * 8][16 + l16] = f2bf(s1[r]);
    }
    v16bf pa = load_a_frag(pw[l16], half);

    // O += P @ V  over all 256 channels (16 tiles)
#pragma unroll
    for (int ct = 0; ct < 16; ++ct) {
      v16bf vb = load_b_frag(ldsV[ct * 16 + l16], half);
      acc[ct] = wmma_bf16(pa, vb, acc[ct]);
    }
    wg_barrier_ds();
  }

  // epilogue: normalize, gamma*O + x, coalesced via LDS transpose
  const float g = gamma[0];
  float inv[8];
#pragma unroll
  for (int r = 0; r < 8; ++r) inv[r] = 1.0f / ssum[r];

  for (int cc = 0; cc < 16; cc += 2) {     // 32 channels per pass
    wg_barrier_ds();
#pragma unroll
    for (int q2 = 0; q2 < 2; ++q2) {
      int ct = cc + q2;
#pragma unroll
      for (int r = 0; r < 8; ++r) {
        int cLoc = q2 * 16 + l16;
        int nLoc = wid * 16 + r + half * 8;
        ldsO[cLoc][nLoc] = acc[ct][r] * inv[r];
      }
    }
    wg_barrier_ds();
#pragma unroll
    for (int j = 0; j < 16; ++j) {
      int idx  = tid + 256 * j;            // 32*128
      int cLoc = idx >> 7, nl = idx & 127;
      int c    = cc * 16 + cLoc;
      size_t gi = ((size_t)b * CCH + c) * NTOK + n0 + nl;
      out[gi] = g * ldsO[cLoc][nl] + x[gi];
    }
  }
}

// ---------------------------------------------------------------------------
extern "C" void kernel_launch(void* const* d_in, const int* in_sizes, int n_in,
                              void* d_out, int out_size, void* d_ws, size_t ws_size,
                              hipStream_t stream) {
  const float* x     = (const float*)d_in[0];
  const float* wq    = (const float*)d_in[1];
  const float* bq    = (const float*)d_in[2];
  const float* wk    = (const float*)d_in[3];
  const float* bk    = (const float*)d_in[4];
  const float* wv    = (const float*)d_in[5];
  const float* bv    = (const float*)d_in[6];
  const float* gamma = (const float*)d_in[7];
  float* out = (float*)d_out;

  char* ws = (char*)d_ws;
  const size_t Y_BYTES = (size_t)BN * OALL * NTOK * sizeof(unsigned short); // ~40 MB
  const size_t W_BYTES = (size_t)OALL * CCH * sizeof(unsigned short);
  unsigned short* Yw      = (unsigned short*)ws;
  unsigned short* Wall    = (unsigned short*)(ws + Y_BYTES);
  float*          biasAll = (float*)(ws + Y_BYTES + W_BYTES);

  pack_kernel<<<OALL, CCH, 0, stream>>>(wq, bq, wk, bk, wv, bv, Wall, biasAll);
  proj_kernel<<<dim3(NTOK / 64, OALL / 64, BN), 256, 0, stream>>>(x, Wall, biasAll, Yw);
  attn_kernel<<<dim3(NTOK / 128, BN), 256, 0, stream>>>(Yw, x, gamma, out);
  (void)in_sizes; (void)n_in; (void)out_size; (void)ws_size;
}
